// MultiHeadSelfAttention_81088982548768
// MI455X (gfx1250) — compile-verified, matrix operand order checked by reading, not run on hardware
//
#include <hip/hip_runtime.h>
#include <hip/hip_bf16.h>
#include <math.h>

typedef __bf16 bf16_t;
typedef __attribute__((ext_vector_type(16))) __bf16 v16bf;
typedef __attribute__((ext_vector_type(8)))  float  v8f;

// ---------------------------------------------------------------------------
// fp32 -> bf16 convert (grid-stride, 4 elems/thread)
// ---------------------------------------------------------------------------
__global__ void cvt_f32_bf16(const float* __restrict__ src,
                             bf16_t* __restrict__ dst, long n) {
  long i = ((long)blockIdx.x * blockDim.x + threadIdx.x) * 4;
  long stride = (long)gridDim.x * blockDim.x * 4;
  for (; i < n; i += stride) {
    float4 v = *(const float4*)(src + i);
    bf16_t o[4];
    o[0] = (bf16_t)v.x; o[1] = (bf16_t)v.y;
    o[2] = (bf16_t)v.z; o[3] = (bf16_t)v.w;
    *(uint2*)(dst + i) = *(uint2*)o;
  }
}

// ---------------------------------------------------------------------------
// CDNA5 async global->LDS staging (ASYNCcnt-tracked, bypasses VGPRs)
// ---------------------------------------------------------------------------
__device__ __forceinline__ void async_b128(unsigned lds_off,
                                           const bf16_t* g) {
  asm volatile("global_load_async_to_lds_b128 %0, %1, off"
               :: "v"(lds_off), "v"((unsigned long long)(uintptr_t)g)
               : "memory");
}

// ---------------------------------------------------------------------------
// Generic NT bf16 GEMM: C[M,N] = alpha * (A[M,K] x B[N,K]^T) (+ bias) (+ beta*C)
// Block tile 128(M) x 256(N), 256 threads = 8 wave32 waves (2 along M x 4
// along N), each wave a 64x64 tile = 4x4 grid of 16x16x32 WMMA per K-step.
// Global->LDS staging uses global_load_async_to_lds_b128, double buffered.
// Exactly one of Cb (bf16 out) / Cf (f32 out) is non-null.
// ---------------------------------------------------------------------------
__global__ __launch_bounds__(256)
void gemm_nt_bf16(const bf16_t* __restrict__ A, long lda,
                  const bf16_t* __restrict__ B, long ldb,
                  int K,
                  bf16_t* __restrict__ Cb, float* __restrict__ Cf, long ldc,
                  float alpha, int beta, const float* __restrict__ bias) {
  __shared__ bf16_t ldsA[2][128 * 32];   // 8 KiB per buffer
  __shared__ bf16_t ldsB[2][256 * 32];   // 16 KiB per buffer

  const int t    = threadIdx.x;
  const int lane = t & 31;
  const int wave = t >> 5;
  const int wm   = wave >> 2;  // 0..1 along M
  const int wn   = wave & 3;   // 0..3 along N
  const long blockM = (long)blockIdx.x * 128;
  const long blockN = (long)blockIdx.y * 256;

  // Staging ownership: A tile 128x32 -> 2 b128/thread, B tile 256x32 -> 4.
  const int rowA = t >> 1;
  const int colA = (t & 1) * 16;
  const int rowB = t;

  const bf16_t* Ag = A + (blockM + rowA) * lda + colA;
  const bf16_t* Bg = B + (blockN + rowB) * ldb;

  unsigned ldsAoff[2], ldsBoff[2];
  ldsAoff[0] = (unsigned)(uintptr_t)&ldsA[0][rowA * 32 + colA];
  ldsAoff[1] = (unsigned)(uintptr_t)&ldsA[1][rowA * 32 + colA];
  ldsBoff[0] = (unsigned)(uintptr_t)&ldsB[0][rowB * 32];
  ldsBoff[1] = (unsigned)(uintptr_t)&ldsB[1][rowB * 32];

  auto issue_stage = [&](int buf, int kt) {
    const bf16_t* ag = Ag + (long)kt * 32;
    const bf16_t* bg = Bg + (long)kt * 32;
    const unsigned la = ldsAoff[buf];
    const unsigned lb = ldsBoff[buf];
    async_b128(la,      ag);
    async_b128(la + 16, ag + 8);
    async_b128(lb,      bg);
    async_b128(lb + 16, bg + 8);
    async_b128(lb + 32, bg + 16);
    async_b128(lb + 48, bg + 24);
  };

  v8f acc[4][4];
#pragma unroll
  for (int i = 0; i < 4; ++i)
#pragma unroll
    for (int j = 0; j < 4; ++j) acc[i][j] = {};

  const int nk = K >> 5;   // K-steps of 32

  issue_stage(0, 0);

  for (int kt = 0; kt < nk; ++kt) {
    const int cur = kt & 1;
    if (kt + 1 < nk) {
      issue_stage(cur ^ 1, kt + 1);
      // Wait until only the 6 just-issued (next-stage) transfers remain.
      asm volatile("s_wait_asynccnt %0" :: "i"(6) : "memory");
    } else {
      asm volatile("s_wait_asynccnt %0" :: "i"(0) : "memory");
    }
    __syncthreads();   // all waves' current-stage tiles are in LDS

    const bf16_t* sA = ldsA[cur];
    const bf16_t* sB = ldsB[cur];
    v16bf aF[4], bF[4];
#pragma unroll
    for (int i = 0; i < 4; ++i)
      aF[i] = *(const v16bf*)&sA[(wm * 64 + i * 16 + (lane & 15)) * 32 +
                                 (lane >> 4) * 16];
#pragma unroll
    for (int j = 0; j < 4; ++j)
      bF[j] = *(const v16bf*)&sB[(wn * 64 + j * 16 + (lane & 15)) * 32 +
                                 (lane >> 4) * 16];

#pragma unroll
    for (int i = 0; i < 4; ++i)
#pragma unroll
      for (int j = 0; j < 4; ++j)
        acc[i][j] = __builtin_amdgcn_wmma_f32_16x16x32_bf16(
            false, aF[i], false, bF[j], (short)0, acc[i][j], false, false);

    __syncthreads();   // everyone done reading buf[cur]; it may be refilled
  }

  // Epilogue per ISA C/D layout: VGPR r -> row = r + (lane>>4)*8, col = lane&15
  const long baseM = blockM + wm * 64;
  const long baseN = blockN + wn * 64;
#pragma unroll
  for (int i = 0; i < 4; ++i) {
#pragma unroll
    for (int j = 0; j < 4; ++j) {
      const long col = baseN + j * 16 + (lane & 15);
#pragma unroll
      for (int r = 0; r < 8; ++r) {
        const long row = baseM + i * 16 + (lane >> 4) * 8 + r;
        const long idx = row * ldc + col;
        float v = acc[i][j][r] * alpha;
        if (Cb) {
          Cb[idx] = (bf16_t)v;
        } else {
          if (bias) v += bias[col];
          if (beta) v += Cf[idx];
          Cf[idx] = v;
        }
      }
    }
  }
}

// ---------------------------------------------------------------------------
// bf16 transpose: dst[c][r] = src[r][c]; 64x64 tiles via LDS.
// grid = (cols/64, rows/64)
// ---------------------------------------------------------------------------
__global__ __launch_bounds__(256)
void transpose_bf16(const bf16_t* __restrict__ src, bf16_t* __restrict__ dst,
                    int rows, int cols) {
  __shared__ bf16_t tile[64][72];
  const long r0 = (long)blockIdx.y * 64;
  const long c0 = (long)blockIdx.x * 64;
  const int t  = threadIdx.x;
  const int tr = t >> 2;          // 0..63
  const int tc = (t & 3) * 16;    // 0,16,32,48

  const bf16_t* sp = src + (r0 + tr) * cols + c0 + tc;
  uint4 v0 = *(const uint4*)sp;
  uint4 v1 = *(const uint4*)(sp + 8);
  *(uint4*)&tile[tr][tc]     = v0;
  *(uint4*)&tile[tr][tc + 8] = v1;
  __syncthreads();

  bf16_t o[16];
#pragma unroll
  for (int u = 0; u < 16; ++u) o[u] = tile[tc + u][tr];
  bf16_t* dp = dst + (c0 + tr) * rows + r0 + tc;
  *(uint4*)dp       = *(uint4*)&o[0];
  *(uint4*)(dp + 8) = *(uint4*)&o[8];
}

// ---------------------------------------------------------------------------
// Column softmax: w[n][m] = softmax over n of S[n][m] (query-axis softmax).
// Block: 16 cols x 16 row-groups. grid.x = N/16.
// ---------------------------------------------------------------------------
__global__ __launch_bounds__(256)
void softmax_cols(const float* __restrict__ S, bf16_t* __restrict__ W, int N) {
  __shared__ float red[16][16];
  const int t  = threadIdx.x;
  const int c  = t & 15;
  const int rg = t >> 4;
  const long m = (long)blockIdx.x * 16 + c;

  float mx = -INFINITY;
  for (int n = rg; n < N; n += 16) mx = fmaxf(mx, S[(long)n * N + m]);
  red[rg][c] = mx;
  __syncthreads();
  if (rg == 0) {
    float v = red[0][c];
    for (int k = 1; k < 16; ++k) v = fmaxf(v, red[k][c]);
    red[0][c] = v;
  }
  __syncthreads();
  mx = red[0][c];
  __syncthreads();

  float sm = 0.f;
  for (int n = rg; n < N; n += 16) sm += __expf(S[(long)n * N + m] - mx);
  red[rg][c] = sm;
  __syncthreads();
  if (rg == 0) {
    float v = 0.f;
    for (int k = 0; k < 16; ++k) v += red[k][c];
    red[0][c] = v;
  }
  __syncthreads();
  const float inv = 1.f / red[0][c];

  for (int n = rg; n < N; n += 16) {
    const long idx = (long)n * N + m;
    W[idx] = (bf16_t)(__expf(S[idx] - mx) * inv);
  }
}

// ---------------------------------------------------------------------------
// Host-side orchestration
// ---------------------------------------------------------------------------
extern "C" void kernel_launch(void* const* d_in, const int* in_sizes, int n_in,
                              void* d_out, int out_size, void* d_ws,
                              size_t ws_size, hipStream_t stream) {
  const float* x    = (const float*)d_in[0];
  const float* Wq   = (const float*)d_in[1];
  const float* Wk   = (const float*)d_in[2];
  const float* Wv   = (const float*)d_in[3];
  const float* fc_w = (const float*)d_in[4];
  const float* fc_b = (const float*)d_in[5];
  float* out = (float*)d_out;

  const long E = 2048, N = 2048;
  const int  B = 4, H = 3;
  const long NE = N * E;          // 4M
  const long EE = E * E;          // 4M
  const long BNE = (long)B * NE;  // 16M
  const float scale = 0.022097086912079608f;  // 1/sqrt(2048)

  char*  ws  = (char*)d_ws;
  size_t off = 0;
  auto alloc = [&](size_t bytes) -> void* {
    void* p = ws + off;
    off += (bytes + 255) & ~(size_t)255;
    return p;
  };

  bf16_t* xb    = (bf16_t*)alloc(BNE * 2);
  bf16_t* Wqb   = (bf16_t*)alloc((size_t)H * EE * 2);
  bf16_t* Wkb   = (bf16_t*)alloc((size_t)H * EE * 2);
  bf16_t* Wvb   = (bf16_t*)alloc((size_t)H * EE * 2);
  bf16_t* fcwb  = (bf16_t*)alloc(EE * 2);
  bf16_t* Qb    = (bf16_t*)alloc(NE * 2);
  bf16_t* Kb    = (bf16_t*)alloc(NE * 2);
  bf16_t* Vb    = (bf16_t*)alloc(NE * 2);
  bf16_t* Vtb   = (bf16_t*)alloc(NE * 2);
  float*  score = (float*)alloc((size_t)N * N * 4);
  bf16_t* wb    = (bf16_t*)alloc((size_t)N * N * 2);
  float*  meanf = (float*)alloc(BNE * 4);
  bf16_t* meanb = (bf16_t*)alloc(BNE * 2);

  auto cvt = [&](const float* s, bf16_t* d, long n) {
    int blocks = (int)(n / (4 * 256));
    cvt_f32_bf16<<<blocks, 256, 0, stream>>>(s, d, n);
  };
  cvt(x, xb, BNE);
  cvt(Wq, Wqb, (long)H * EE);
  cvt(Wk, Wkb, (long)H * EE);
  cvt(Wv, Wvb, (long)H * EE);
  cvt(fc_w, fcwb, EE);

  auto gemm = [&](const bf16_t* A, long lda, const bf16_t* Bm, long ldb,
                  int M, int Nc, int K, bf16_t* Cb, float* Cf, long ldc,
                  float alpha, int beta, const float* bias) {
    dim3 grid(M / 128, Nc / 256);
    gemm_nt_bf16<<<grid, 256, 0, stream>>>(A, lda, Bm, ldb, K, Cb, Cf, ldc,
                                           alpha, beta, bias);
  };

  for (int b = 0; b < B; ++b) {
    for (int h = 0; h < H; ++h) {
      const bf16_t* xbb = xb + (long)b * NE;
      // Q/K/V = x @ W[h].T  (NT, bf16 out)
      gemm(xbb, E, Wqb + (long)h * EE, E, (int)N, (int)E, (int)E,
           Qb, nullptr, E, 1.f, 0, nullptr);
      gemm(xbb, E, Wkb + (long)h * EE, E, (int)N, (int)E, (int)E,
           Kb, nullptr, E, 1.f, 0, nullptr);
      gemm(xbb, E, Wvb + (long)h * EE, E, (int)N, (int)E, (int)E,
           Vb, nullptr, E, 1.f, 0, nullptr);
      // V^T so attn GEMM is also NT
      {
        dim3 grid(E / 64, N / 64);
        transpose_bf16<<<grid, 256, 0, stream>>>(Vb, Vtb, (int)N, (int)E);
      }
      // scores = scale * Q @ K^T  (f32 out)
      gemm(Qb, E, Kb, E, (int)N, (int)N, (int)E,
           nullptr, score, N, scale, 0, nullptr);
      // softmax over the query axis (columns of score viewed [n][m])
      softmax_cols<<<(int)(N / 16), 256, 0, stream>>>(score, wb, (int)N);
      // mean[b] (+)= (1/3) * w @ V   == w @ Vt^T  (NT, f32 accumulate)
      gemm(wb, N, Vtb, N, (int)N, (int)E, (int)N,
           nullptr, meanf + (long)b * NE, E, 1.f / 3.f, (h > 0) ? 1 : 0,
           nullptr);
    }
  }

  // mean -> bf16, then out = mean @ fc_w.T + fc_b
  cvt(meanf, meanb, BNE);
  gemm(meanb, E, fcwb, E, (int)(B * N), (int)E, (int)E,
       nullptr, out, E, 1.f, 0, fc_b);
}
